// LSTMbiDir_8392366096811
// MI455X (gfx1250) — matrix operand order repaired and checked by reading, not compiled, not run on hardware
//
#include <hip/hip_runtime.h>
#include <hip/hip_bf16.h>

// ---------------------------------------------------------------------------
// Bidirectional LSTM for MI455X (gfx1250, wave32, WMMA).
//   - bf16 WMMA 16x16x32 for gate GEMMs (K = E+H = 384 -> 12 k-tiles).
//   - Persistent workgroup per (32-row batch tile x direction); c/h state in
//     VGPRs across the whole T=1024 scan; h re-packed into LDS A-fragments.
//   - Weights pre-packed bf16 so each lane's B-fragment is one contiguous
//     32B load streamed from L2 every step. An opaque asm-carried offset
//     keeps these loads loop-variant (prevents LICM hoist -> scratch spill).
// ---------------------------------------------------------------------------

typedef __bf16 bf16_t;
typedef __attribute__((ext_vector_type(16))) __bf16 v16bf;
typedef __attribute__((ext_vector_type(8)))  float  v8f;

#define VV    30000
#define EE    128
#define HH    256
#define BB    512
#define TT    1024
#define KDIM  384            // EE + HH
#define KT    12             // KDIM / 32
#define MATBF ((size_t)KT * 16 * 32 * 16)   // bf16 elems per packed matrix = 98304
#define KTBF  ((size_t)16 * 32 * 16)        // bf16 elems per (mat,kt) slab  = 8192

// Workspace byte offsets
#define WS_WPACK 0
#define WS_EMB   (8UL * KDIM * HH * 2)                 //  1,572,864
#define WS_MASK  (WS_EMB + (size_t)VV * EE * 2)        // +7,680,000
#define WS_HCAT  (WS_MASK + (size_t)BB * TT)           // +  524,288

struct Ptrs8 { const float* p[8]; };

__device__ __forceinline__ float sigmoidf_(float x) {
    return 1.0f / (1.0f + __expf(-x));
}

// --------------------------- prep: weight packing ---------------------------
__global__ __launch_bounds__(256) void wpack_kernel(Ptrs8 w, bf16_t* __restrict__ wpack) {
    int gid = blockIdx.x * 256 + threadIdx.x;           // 8*384*256 total
    int mat = gid / (KDIM * HH);
    int rem = gid % (KDIM * HH);
    int k = rem >> 8;                                   // 0..383
    int n = rem & 255;                                  // 0..255
    float v = w.p[mat][k * HH + n];
    int kt = k >> 5, ko = k & 31;
    int lane = (n & 15) + (((ko >> 3) & 1) << 4);
    int j    = (ko & 7) + ((ko >> 4) << 3);
    int ntile = n >> 4;
    wpack[(((size_t)mat * KT + kt) * 16 + ntile) * 512 + lane * 16 + j] = (bf16_t)v;
}

// --------------------------- prep: embedding -> bf16 ------------------------
__global__ __launch_bounds__(256) void embconv_kernel(const float* __restrict__ emb,
                                                      bf16_t* __restrict__ embBf) {
    int gid = blockIdx.x * 256 + threadIdx.x;           // VV*EE total
    embBf[gid] = (bf16_t)emb[gid];
}

// --------------------------- prep: padding mask ----------------------------
__global__ __launch_bounds__(256) void mask_kernel(const long long* __restrict__ data,
                                                   const float* __restrict__ emb,
                                                   unsigned char* __restrict__ mask) {
    int gid = blockIdx.x * 256 + threadIdx.x;           // BB*TT total
    int b = gid >> 10;
    long long il = data[(size_t)b * TT + (TT - 1)];
    long long it = data[gid];
    const float4* a = (const float4*)(emb + (size_t)it * EE);
    const float4* c = (const float4*)(emb + (size_t)il * EE);
    int any = 0;
    #pragma unroll 8
    for (int i = 0; i < EE / 4; ++i) {
        float4 x = a[i], y = c[i];
        any |= (x.x != y.x) | (x.y != y.y) | (x.z != y.z) | (x.w != y.w);
    }
    mask[gid] = (unsigned char)any;
}

// --------------------------- main recurrent kernel -------------------------
// grid.x = (BB/32) * 2 dirs = 32 blocks of 512 threads (16 wave32).
// Wave w: mtile = w>>3 (16 rows), ngrp = w&7 (32 of 256 h-cols), all 4 gates.
__global__ __launch_bounds__(512) void lstm_scan_kernel(
    const long long*     __restrict__ data,
    const bf16_t*        __restrict__ embBf,   // [VV][EE]
    const unsigned char* __restrict__ mask,    // [BB][TT]
    const bf16_t*        __restrict__ wpack,   // [8][KT][16][32][16]
    Ptrs8 biases,
    float*               __restrict__ hcat)    // [BB][2*HH]
{
    __shared__ v16bf ldsA[2][KT][32];          // xh A-fragments, 24 KB
    __shared__ float ldsMask[2][32];           // double-buffered step mask

    const int tid   = threadIdx.x;
    const int lane  = tid & 31;
    const int wave  = tid >> 5;
    const int mtile = wave >> 3;               // 0..1
    const int ngrp  = wave & 7;                // 0..7 (32 columns each)
    const int dir   = blockIdx.x & 1;
    const int bbase = (blockIdx.x >> 1) * 32;

    bf16_t*       ldsAh = (bf16_t*)&ldsA[0][0][0];
    unsigned int* ldsAu = (unsigned int*)ldsAh;

    // Zero the fragment buffer (h region must start at 0 == initial state).
    for (int i = tid; i < 2 * KT * 32 * 8; i += 512) ldsAu[i] = 0u;

    // Bias scalars: within a 16x16 D-tile each thread's accum column is
    // N = lane%16, so one scalar per (gate, ntile).
    float biasv[4][2];
    const int ncol0 = ngrp * 32 + (lane & 15);
    #pragma unroll
    for (int g = 0; g < 4; ++g) {
        const float* bp = biases.p[dir * 4 + g];
        #pragma unroll
        for (int nt = 0; nt < 2; ++nt) biasv[g][nt] = bp[ncol0 + nt * 16];
    }

    float creg[2][8], hreg[2][8];              // persistent cell/hidden state
    #pragma unroll
    for (int nt = 0; nt < 2; ++nt)
        #pragma unroll
        for (int r = 0; r < 8; ++r) { creg[nt][r] = 0.0f; hreg[nt][r] = 0.0f; }

    __syncthreads();

    const int grow = tid >> 4;                 // gather: row 0..31
    const int gcol = tid & 15;                 // gather: dword slot group
    const bf16_t* wbase = wpack + (size_t)(dir * 4) * MATBF;
    const int hi = lane >> 4;

    // Opaque offset: formally loop-variant so the compiler cannot hoist the
    // (address-invariant) weight loads out of the t-loop and spill them.
    unsigned int woff = 0;

    for (int t = 0; t < TT; ++t) {
        asm volatile("" : "+s"(woff));
        const bf16_t* wt = wbase + woff;
        const int teff = dir ? (TT - 1 - t) : t;

        // ---- (a) gather x_t into A-fragments (k-tiles 0..3) ----
        {
            const long long idx = data[(size_t)(bbase + grow) * TT + teff];
            const unsigned int* src = (const unsigned int*)(embBf + (size_t)idx * EE);
            const int mt = grow >> 4, ml = grow & 15;
            #pragma unroll
            for (int i = 0; i < 4; ++i) {
                int dw = gcol * 4 + i;                  // 0..63 (2 bf16 each)
                int k  = dw * 2;
                int kt = k >> 5, ko = k & 31, grp = ko >> 3;
                int dl = ml + ((grp & 1) << 4);
                int jd = ((ko & 7) >> 1) + ((grp >> 1) << 2);
                ldsAu[(((mt * KT + kt) * 32 + dl) << 3) + jd] = src[dw];
            }
            if (tid < 32)
                ldsMask[t & 1][tid] =
                    mask[(size_t)(bbase + tid) * TT + teff] ? 1.0f : 0.0f;
        }
        __syncthreads();

        // ---- (b) GEMM: xh(16x384) @ W(384x32) for 4 gates ----
        v8f acc[4][2];
        #pragma unroll
        for (int g = 0; g < 4; ++g)
            #pragma unroll
            for (int nt = 0; nt < 2; ++nt) {
                v8f a;
                #pragma unroll
                for (int r = 0; r < 8; ++r) a[r] = biasv[g][nt];
                acc[g][nt] = a;
            }

        for (int kt = 0; kt < KT; ++kt) {
            const v16bf afrag = ldsA[mtile][kt][lane];   // 32B from LDS
            if (kt + 1 < KT)                              // warm next k-slab
                __builtin_prefetch(wt + (size_t)(kt + 1) * KTBF, 0, 3);
            #pragma unroll
            for (int g = 0; g < 4; ++g) {
                const bf16_t* wg = wt + ((size_t)g * KT + kt) * KTBF;
                #pragma unroll
                for (int nt = 0; nt < 2; ++nt) {
                    const v16bf bfrag =
                        *(const v16bf*)(wg + ((ngrp * 2 + nt) * 32 + lane) * 16);
                    acc[g][nt] = __builtin_amdgcn_wmma_f32_16x16x32_bf16(
                        false, afrag, false, bfrag, (short)0, acc[g][nt],
                        false, false);
                }
            }
        }
        __syncthreads();

        // ---- (c) gate math in registers; scatter new h into A-fragments ----
        #pragma unroll
        for (int nt = 0; nt < 2; ++nt) {
            const int ncol  = ngrp * 32 + nt * 16 + (lane & 15);
            const int kglob = EE + ncol;                 // h occupies K 128..383
            const int kt2 = kglob >> 5, ko = kglob & 31, grp = ko >> 3;
            const int jj  = (ko & 7) + ((grp >> 1) << 3);
            #pragma unroll
            for (int r = 0; r < 8; ++r) {
                const int mrow = mtile * 16 + r + (hi << 3);
                const float mk = ldsMask[t & 1][mrow];
                float f  = sigmoidf_(acc[0][nt][r]);
                float ii = sigmoidf_(acc[1][nt][r]);
                float o  = sigmoidf_(acc[2][nt][r]);
                float g  = tanhf(acc[3][nt][r]);
                float c0 = creg[nt][r];
                float c2 = mk * (c0 * f + ii * g) + (1.0f - mk) * c0;
                float h2 = mk * (o * tanhf(c2)) + (1.0f - mk) * hreg[nt][r];
                creg[nt][r] = c2;
                hreg[nt][r] = h2;
                const int dl = (mrow & 15) + ((grp & 1) << 4);
                ldsAh[((((mrow >> 4) * KT + kt2) * 32 + dl) << 4) + jj] = (bf16_t)h2;
            }
        }
        // No barrier needed: next (a) writes only x-region / other mask buffer.
    }

    // ---- final hidden state -> hcat[b][dir*HH + n] ----
    #pragma unroll
    for (int nt = 0; nt < 2; ++nt) {
        const int ncol = ngrp * 32 + nt * 16 + (lane & 15);
        #pragma unroll
        for (int r = 0; r < 8; ++r) {
            const int mrow = mtile * 16 + r + (hi << 3);
            hcat[(size_t)(bbase + mrow) * (2 * HH) + dir * HH + ncol] = hreg[nt][r];
        }
    }
}

// --------------------------- output head ----------------------------------
__global__ __launch_bounds__(256) void out_kernel(const float* __restrict__ hcat,
                                                  const float* __restrict__ Wo,
                                                  const float* __restrict__ bo,
                                                  float* __restrict__ out) {
    int b = blockIdx.x * 256 + threadIdx.x;
    if (b >= BB) return;
    float s = bo[0];
    const float* hr = hcat + (size_t)b * (2 * HH);
    #pragma unroll 8
    for (int j = 0; j < 2 * HH; ++j) s += hr[j] * Wo[j];
    out[b] = sigmoidf_(s);
}

// --------------------------- launch ----------------------------------------
extern "C" void kernel_launch(void* const* d_in, const int* in_sizes, int n_in,
                              void* d_out, int out_size, void* d_ws, size_t ws_size,
                              hipStream_t stream) {
    const long long* data = (const long long*)d_in[0];
    const float*     emb  = (const float*)d_in[1];

    char* ws = (char*)d_ws;
    bf16_t*        wpackp = (bf16_t*)(ws + WS_WPACK);
    bf16_t*        embBf  = (bf16_t*)(ws + WS_EMB);
    unsigned char* maskp  = (unsigned char*)(ws + WS_MASK);
    float*         hcat   = (float*)(ws + WS_HCAT);

    Ptrs8 wp, bp;
    for (int i = 0; i < 8; ++i) {
        wp.p[i] = (const float*)d_in[2 + i * 2];   // Wfh1,Wih1,Woh1,Wch1,Wfh2,...
        bp.p[i] = (const float*)d_in[3 + i * 2];   // bfh1,bih1,boh1,bch1,bfh2,...
    }

    wpack_kernel  <<<(8 * KDIM * HH) / 256, 256, 0, stream>>>(wp, wpackp);
    embconv_kernel<<<(VV * EE) / 256,       256, 0, stream>>>(emb, embBf);
    mask_kernel   <<<(BB * TT) / 256,       256, 0, stream>>>(data, emb, maskp);

    lstm_scan_kernel<<<(BB / 32) * 2, 512, 0, stream>>>(data, embBf, maskp,
                                                        wpackp, bp, hcat);

    out_kernel<<<(BB + 255) / 256, 256, 0, stream>>>(
        hcat, (const float*)d_in[18], (const float*)d_in[19], (float*)d_out);
}
